// SpectralConv1d_3977139716490
// MI455X (gfx1250) — matrix operand order repaired and checked by reading, not compiled, not run on hardware
//
#include <hip/hip_runtime.h>

// ---------------------------------------------------------------------------
// SpectralConv1d on MI455X (gfx1250, wave32):
//   fwd FHT (LDS-resident radix-2 DIT) -> per-mode bf16 WMMA GEMMs -> inv FHT
// B=32, CIN=64, COUT=64, N=8192, MODES=1024
// ---------------------------------------------------------------------------

#define B_     32
#define CIN_   64
#define COUT_  64
#define N_     8192
#define MODES_ 1024
#define LOGN_  13

typedef __attribute__((ext_vector_type(16))) __bf16 v16bf;
typedef __attribute__((ext_vector_type(8)))  __bf16 bf16x8;
typedef __attribute__((ext_vector_type(8)))  float  v8f;

// ---------------------------------------------------------------------------
// 0) cas twiddle table: tab[L + k] = cos(pi*k/L) + sin(pi*k/L), k in [0,L)
//    stage L occupies [L, 2L) -> 8192 floats total, L2-resident forever.
// ---------------------------------------------------------------------------
__global__ void __launch_bounds__(256)
cas_init_kernel(float* __restrict__ tab)
{
    int t = blockIdx.x * 256 + threadIdx.x;           // 0..8191
    if (t == 0) { tab[0] = 1.0f; return; }
    int lg = 31 - __clz(t);
    int L  = 1 << lg;
    int k  = t - L;
    float th = 3.14159265358979323846f * (float)k / (float)L;
    tab[t] = __cosf(th) + __sinf(th);
}

// ---------------------------------------------------------------------------
// 1) weight prep: weven/wodd in bf16, layout (m, o, i) so a WMMA B-lane
//    (column o) reads its K(=i) slice contiguously.
//    weven[i,o,m] = 0.5*(w[i,o,m] + w[i,o,(1024-m)%1024]); wodd = 0.5*(diff)
// ---------------------------------------------------------------------------
__global__ void __launch_bounds__(256)
prep_w_kernel(const float* __restrict__ w,
              __bf16* __restrict__ we, __bf16* __restrict__ wo)
{
    unsigned t = blockIdx.x * 256u + threadIdx.x;     // < 1024*64*64
    int i  = t & 63;
    int o  = (t >> 6) & 63;
    int m  = t >> 12;
    int mf = (MODES_ - m) & (MODES_ - 1);
    float a = w[((size_t)i * COUT_ + o) * MODES_ + m];
    float b = w[((size_t)i * COUT_ + o) * MODES_ + mf];
    size_t oi = ((size_t)m * COUT_ + o) * CIN_ + i;
    we[oi] = (__bf16)(0.5f * (a + b));
    wo[oi] = (__bf16)(0.5f * (a - b));
}

// ---------------------------------------------------------------------------
// 2) forward FHT: one 8192-pt row per workgroup, 32KB LDS, bit-reversed
//    load then 13 table-driven butterfly stages. Emit first 1024 modes as
//    bf16 into xm[(m, b, i)] = xm[m*2048 + blockIdx.x].
// ---------------------------------------------------------------------------
__global__ void __launch_bounds__(256)
fht_fwd_kernel(const float* __restrict__ x, const float* __restrict__ tab,
               __bf16* __restrict__ xm)
{
    __shared__ float s[N_];
    const int tid = threadIdx.x;
    const size_t rowbase = (size_t)blockIdx.x * N_;

    // bit-reversed gather (contiguous LDS writes; global scatter hits L2)
    for (int j = tid; j < N_; j += 256)
        s[j] = x[rowbase + (__brev((unsigned)j) >> (32 - LOGN_))];
    __syncthreads();

    for (int L = 1; L < N_; L <<= 1) {
        for (int t = tid; t < (N_ / 2); t += 256) {
            int k  = t & (L - 1);
            int i0 = ((t - k) << 1) + k;              // block*2L + k
            int i1 = i0 + L;
            float cas = tab[L + k];
            float E = s[i0], O = s[i1];
            float tt = cas * O;
            s[i0] = E + tt;
            s[i1] = E - tt;
        }
        __syncthreads();
    }

    for (int m = tid; m < MODES_; m += 256)
        xm[(size_t)m * (B_ * CIN_) + blockIdx.x] = (__bf16)s[m];
}

// ---------------------------------------------------------------------------
// 3) per-mode GEMM on v_wmma_f32_16x16x32_bf16.
//    z_m(32x64) = xm_m(32x64) @ We_m(64x64) + xm_{-m}(32x64) @ Wo_m(64x64)
//    1 block = 1 mode, 8 waves = 2x4 tile grid of 16x16 outputs,
//    4 chained WMMAs per wave (even/odd x two K=32 chunks).
// ---------------------------------------------------------------------------
__device__ __forceinline__ v16bf load_a_frag(const __bf16* p)
{
    // A 16x32 bf16 layout: elems 0-15 = K klo..klo+15 (contiguous)
    bf16x8 lo = *(const bf16x8*)p;
    bf16x8 hi = *(const bf16x8*)(p + 8);
    v16bf r;
#pragma unroll
    for (int e = 0; e < 8; ++e) { r[e] = lo[e]; r[e + 8] = hi[e]; }
    return r;
}

__global__ void __launch_bounds__(256)
mode_gemm_kernel(const __bf16* __restrict__ xm, const __bf16* __restrict__ we,
                 const __bf16* __restrict__ wo, float* __restrict__ z)
{
    const int m    = blockIdx.x;
    const int mf   = (MODES_ - m) & (MODES_ - 1);
    const int lane = threadIdx.x & 31;
    const int wave = threadIdx.x >> 5;
    const int mt   = wave >> 2;                       // 0..1  (rows of B_)
    const int nt   = wave & 3;                        // 0..3  (cols of COUT_)
    const int lh   = lane >> 4;
    const int ll   = lane & 15;

    const int brow = mt * 16 + ll;                    // A-lane row (batch)
    const int klo  = lh * 16;                         // A-lane K base
    const int ocol = nt * 16 + ll;                    // B-lane column (cout)
    const int kbB  = lh * 16;                         // B-lane K base

    const __bf16* aE = xm + (size_t)m  * (B_ * CIN_) + brow * CIN_ + klo;
    const __bf16* aO = xm + (size_t)mf * (B_ * CIN_) + brow * CIN_ + klo;
    const __bf16* bE = we + ((size_t)m * COUT_ + ocol) * CIN_ + kbB;
    const __bf16* bO = wo + ((size_t)m * COUT_ + ocol) * CIN_ + kbB;

    v8f acc = {};
#pragma unroll
    for (int kc = 0; kc < 2; ++kc) {
        v16bf Ae = load_a_frag(aE + kc * 32);
        v16bf Be = *(const v16bf*)(bE + kc * 32);
        acc = __builtin_amdgcn_wmma_f32_16x16x32_bf16(
                  false, Ae, false, Be, (short)0, acc, false, false);
        v16bf Ao = load_a_frag(aO + kc * 32);
        v16bf Bo = *(const v16bf*)(bO + kc * 32);
        acc = __builtin_amdgcn_wmma_f32_16x16x32_bf16(
                  false, Ao, false, Bo, (short)0, acc, false, false);
    }

    // D layout: VGPR r -> row (lh*8 + r), col ll. Store z[(m, b, o)].
#pragma unroll
    for (int r = 0; r < 8; ++r) {
        int row = mt * 16 + lh * 8 + r;
        z[(size_t)m * (B_ * COUT_) + row * COUT_ + ocol] = acc[r];
    }
}

// ---------------------------------------------------------------------------
// 4) inverse FHT: input nonzero only for modes < 1024. After bit-reversal
//    the nonzeros land at positions 8j, so the first 3 stages collapse to an
//    8-wide broadcast: lds[8j..8j+7] = z[brev10(j)]; start at L=8.
//    Output scaled by 1/N, stored contiguously.
// ---------------------------------------------------------------------------
__global__ void __launch_bounds__(256)
fht_inv_kernel(const float* __restrict__ z, const float* __restrict__ tab,
               float* __restrict__ out)
{
    __shared__ float s[N_];
    const int tid = threadIdx.x;

    for (int j = tid; j < MODES_; j += 256) {
        int src = __brev((unsigned)j) >> 22;          // 10-bit reversal
        float v = z[(size_t)src * (B_ * COUT_) + blockIdx.x];
#pragma unroll
        for (int r = 0; r < 8; ++r) s[j * 8 + r] = v;
    }
    __syncthreads();

    for (int L = 8; L < N_; L <<= 1) {
        for (int t = tid; t < (N_ / 2); t += 256) {
            int k  = t & (L - 1);
            int i0 = ((t - k) << 1) + k;
            int i1 = i0 + L;
            float cas = tab[L + k];
            float E = s[i0], O = s[i1];
            float tt = cas * O;
            s[i0] = E + tt;
            s[i1] = E - tt;
        }
        __syncthreads();
    }

    const size_t rowbase = (size_t)blockIdx.x * N_;
    const float inv_n = 1.0f / (float)N_;
    for (int j = tid; j < N_; j += 256)
        out[rowbase + j] = s[j] * inv_n;
}

// ---------------------------------------------------------------------------
// launch: ws layout (bytes):
//   [0,32K)           cas table (f32 x 8192)
//   [32K, +4MB)       xm   bf16 (MODES, B, CIN)
//   [+4MB, +12MB)     we   bf16 (MODES, COUT, CIN)
//   [+12MB, +20MB)    wo   bf16 (MODES, COUT, CIN)
//   [+20MB, +28MB)    z    f32  (MODES, B, COUT)
// ---------------------------------------------------------------------------
extern "C" void kernel_launch(void* const* d_in, const int* in_sizes, int n_in,
                              void* d_out, int out_size, void* d_ws, size_t ws_size,
                              hipStream_t stream)
{
    (void)in_sizes; (void)n_in; (void)out_size; (void)ws_size;
    const float* x  = (const float*)d_in[0];          // (B, CIN, N) f32
    const float* w1 = (const float*)d_in[1];          // (CIN, COUT, MODES) f32
    float* out = (float*)d_out;                       // (B, COUT, N) f32

    char* ws = (char*)d_ws;
    float*  cas_tab = (float*)ws;
    __bf16* xm = (__bf16*)(ws + (32u << 10));
    __bf16* we = (__bf16*)(ws + (32u << 10) + (4u  << 20));
    __bf16* wo = (__bf16*)(ws + (32u << 10) + (12u << 20));
    float*  z  = (float*) (ws + (32u << 10) + (20u << 20));

    cas_init_kernel <<<N_ / 256, 256, 0, stream>>>(cas_tab);
    prep_w_kernel   <<<(CIN_ * COUT_ * MODES_) / 256, 256, 0, stream>>>(w1, we, wo);
    fht_fwd_kernel  <<<B_ * CIN_,  256, 0, stream>>>(x, cas_tab, xm);
    mode_gemm_kernel<<<MODES_,     256, 0, stream>>>(xm, we, wo, z);
    fht_inv_kernel  <<<B_ * COUT_, 256, 0, stream>>>(z, cas_tab, out);
}
